// PAM_38319698215551
// MI455X (gfx1250) — matrix operand order repaired and checked by reading, not compiled, not run on hardware
//
#include <hip/hip_runtime.h>

typedef __attribute__((ext_vector_type(16))) _Float16 v16h;
typedef __attribute__((ext_vector_type(8)))  _Float16 v8h;
typedef __attribute__((ext_vector_type(8)))  float    v8f;
typedef __attribute__((ext_vector_type(4)))  float    v4f;
typedef __attribute__((ext_vector_type(4)))  int      v4i;

#define BATCH 8
#define NPIX  4096
#define CCH   256
#define C8    32

// workspace layout, in _Float16 units
static constexpr size_t WB_PK  = 0;                         // 256*32   = 8192
static constexpr size_t WC_PK  = 8192;                      // 256*32   = 8192
static constexpr size_t WD_PK  = 16384;                     // 256*256  = 65536
static constexpr size_t BQ_OFF = 81920;                     // 8*4096*32
static constexpr size_t CK_OFF = BQ_OFF + (size_t)BATCH*NPIX*C8;
static constexpr size_t DT_OFF = CK_OFF + (size_t)BATCH*NPIX*C8;   // 8*256*4096

// --- CDNA5 async global->LDS copy (ASYNCcnt-tracked), 16B per lane ----------
// builtin signature (from hipcc diagnostic): (v4i addrspace(1)*, v4i addrspace(3)*, imm, imm)
typedef __attribute__((address_space(1))) v4i gv4i;
typedef __attribute__((address_space(3))) v4i lv4i;

__device__ __forceinline__ void async_cp16(const _Float16* gp, _Float16* lp) {
#if __has_builtin(__builtin_amdgcn_global_load_async_to_lds_b128)
  gv4i* g1 = (gv4i*)(v4i*)gp;   // generic -> as1, pointee v4i
  lv4i* l3 = (lv4i*)(v4i*)lp;   // generic -> as3, pointee v4i
  __builtin_amdgcn_global_load_async_to_lds_b128(g1, l3, 0, 0);
#else
  unsigned loff = (unsigned)(unsigned long long)
      (__attribute__((address_space(3))) void*)lp;  // 32-bit LDS offset
  asm volatile("global_load_async_to_lds_b128 %0, %1, off"
               :: "v"(loff), "v"(gp) : "memory");
#endif
}
__device__ __forceinline__ void wait_async_le9() {
  asm volatile("s_wait_asynccnt 0x9" ::: "memory");
}
__device__ __forceinline__ void wait_async_0() {
  asm volatile("s_wait_asynccnt 0x0" ::: "memory");
}

// ---------------------------------------------------------------------------
// Kernel 0: repack weights (f32 row-major [K][N]) into f16 WMMA B-fragment
// order: frag(nt,kk) = 32 lanes x 16 contiguous halves.
// B-layout: lanes 0-15 hold K=frag_k+0..7,16..23, lanes 16-31 hold K=frag_k+8..15,24..31,
// N = lane&15.
// ---------------------------------------------------------------------------
__global__ void pam_pack_w(const float* __restrict__ Wb,
                           const float* __restrict__ Wc,
                           const float* __restrict__ Wd,
                           _Float16* __restrict__ ws) {
  int idx = blockIdx.x * 256 + threadIdx.x;   // 81920 total
  const float* src; _Float16* dst; int NC; int p;
  if (idx < 8192)        { src = Wb; dst = ws + WB_PK; NC = C8;  p = idx; }
  else if (idx < 16384)  { src = Wc; dst = ws + WC_PK; NC = C8;  p = idx - 8192; }
  else                   { src = Wd; dst = ws + WD_PK; NC = CCH; p = idx - 16384; }
  int nt   = p >> 12;
  int rem  = p & 4095;
  int kk   = rem >> 9;
  int lane = (rem >> 4) & 31;
  int half = rem & 15;
  int k = kk * 32 + ((lane >> 4) << 3) + (half & 7) + ((half >> 3) << 4);
  int n = nt * 16 + (lane & 15);
  dst[p] = (_Float16)src[k * NC + n];
}

// ---------------------------------------------------------------------------
// Kernel 1: projections. One wave = 16 rows of x. Loads 8 A-fragments
// (K=0..255) once, then 20 output 16-col tiles (b:2, c:2, d:16), each
// accumulated over 8 K-step WMMAs. d stored TRANSPOSED [C][N] in f16.
// ---------------------------------------------------------------------------
__global__ void __launch_bounds__(128) pam_proj(const float* __restrict__ x,
                                                const float* __restrict__ bb,
                                                const float* __restrict__ bc,
                                                const float* __restrict__ bd,
                                                _Float16* __restrict__ ws) {
  const int wid  = threadIdx.x >> 5;
  const int lane = threadIdx.x & 31;
  const int hi   = lane >> 4;
  const int ln   = lane & 15;
  const int tile = blockIdx.x * 4 + wid;     // 2048 row tiles
  const int b    = tile >> 8;
  const int n0   = (tile & 255) << 4;

  v16h a[8];
  const float* xrow = x + ((size_t)(b * NPIX + n0 + ln)) * CCH;
  #pragma unroll
  for (int kk = 0; kk < 8; ++kk) {
    int off = kk * 32 + hi * 8;
    v4f f0 = *(const v4f*)(xrow + off);
    v4f f1 = *(const v4f*)(xrow + off + 4);
    v4f f2 = *(const v4f*)(xrow + off + 16);
    v4f f3 = *(const v4f*)(xrow + off + 20);
    #pragma unroll
    for (int i = 0; i < 4; ++i) {
      a[kk][i]      = (_Float16)f0[i];
      a[kk][4 + i]  = (_Float16)f1[i];
      a[kk][8 + i]  = (_Float16)f2[i];
      a[kk][12 + i] = (_Float16)f3[i];
    }
  }

  _Float16* bq = ws + BQ_OFF;
  _Float16* ck = ws + CK_OFF;
  _Float16* dT = ws + DT_OFF;

  for (int nt = 0; nt < 20; ++nt) {
    const _Float16* wp; float bias;
    if (nt < 2)      { wp = ws + WB_PK + (size_t)nt * 4096;       bias = bb[nt * 16 + ln]; }
    else if (nt < 4) { wp = ws + WC_PK + (size_t)(nt - 2) * 4096; bias = bc[(nt - 2) * 16 + ln]; }
    else             { wp = ws + WD_PK + (size_t)(nt - 4) * 4096; bias = bd[(nt - 4) * 16 + ln]; }

    v8f acc;
    #pragma unroll
    for (int v = 0; v < 8; ++v) acc[v] = bias;

    #pragma unroll
    for (int kk = 0; kk < 8; ++kk) {
      v16h w = *(const v16h*)(wp + (size_t)kk * 512 + (size_t)lane * 16);
      acc = __builtin_amdgcn_wmma_f32_16x16x32_f16(false, a[kk], false, w,
                                                   (short)0, acc, false, false);
    }

    if (nt < 4) {
      _Float16* o = (nt < 2) ? bq : ck;
      int ntl = (nt < 2) ? nt : nt - 2;
      #pragma unroll
      for (int v = 0; v < 8; ++v) {
        int row = n0 + v + hi * 8;
        o[((size_t)(b * NPIX + row)) * C8 + ntl * 16 + ln] = (_Float16)acc[v];
      }
    } else {
      int chan = (nt - 4) * 16 + ln;
      _Float16* o = dT + ((size_t)(b * CCH + chan)) * NPIX + n0 + hi * 8;
      v8h pk;
      #pragma unroll
      for (int v = 0; v < 8; ++v) pk[v] = (_Float16)acc[v];
      *(v8h*)o = pk;
    }
  }
}

// ---------------------------------------------------------------------------
// Kernel 2: flash attention with double-buffered async global->LDS staging.
// 4 waves/block share one batch; the key tile (2KB) and transposed value tile
// (16KB) per 32-key step are DMA'd once per block (9 async b128 per wave),
// overlapped with the previous step's 18 WMMAs. ASYNCcnt in-order completion
// lets "s_wait_asynccnt 9" retire the previous tile after issuing the next.
// ---------------------------------------------------------------------------
__global__ void __launch_bounds__(128) pam_attn(const float* __restrict__ x,
                                                const float* __restrict__ gamma,
                                                const _Float16* __restrict__ ws,
                                                float* __restrict__ out) {
  __shared__ __align__(16) _Float16 dTile[2][CCH][32];   // 2 x 16KB
  __shared__ __align__(16) _Float16 ckTile[2][32][32];   // 2 x  2KB
  __shared__ __align__(16) _Float16 ldsP[4][16][40];     // P transpose, 16B rows

  const int wid  = threadIdx.x >> 5;
  const int lane = threadIdx.x & 31;
  const int hi   = lane >> 4;
  const int ln   = lane & 15;
  const int tile = blockIdx.x * 4 + wid;
  const int b    = tile >> 8;          // 256 tiles per batch -> whole block same b
  const int q0   = (tile & 255) << 4;

  const _Float16* bq = ws + BQ_OFF + ((size_t)b * NPIX) * C8;
  const _Float16* ck = ws + CK_OFF + ((size_t)b * NPIX) * C8;
  const _Float16* dT = ws + DT_OFF + ((size_t)b * CCH) * NPIX;

  // cooperative async stage of one 32-key tile into buffer `buf`
  auto issue_tile = [&](int j0n, int buf) {
    const int t = threadIdx.x;
    // keys: rows j0n..j0n+31 are contiguous (2KB): 1 chunk / thread
    async_cp16(ck + (size_t)j0n * C8 + t * 8, &ckTile[buf][0][0] + t * 8);
    // values: 256 rows x 64B at stride 8KB: 8 chunks / thread
    #pragma unroll
    for (int p = 0; p < 8; ++p) {
      int c = t + 128 * p;
      int chan = c >> 2, part = c & 3;
      async_cp16(dT + (size_t)chan * NPIX + j0n + part * 8,
                 &dTile[buf][chan][part * 8]);
    }
  };

  // query A-fragment (16x32, K=C8)
  v16h qa;
  {
    const _Float16* r = bq + (size_t)(q0 + ln) * C8;
    int kb = hi * 16;
    v8h c0 = *(const v8h*)(r + kb);
    v8h c1 = *(const v8h*)(r + kb + 8);
    #pragma unroll
    for (int i = 0; i < 8; ++i) { qa[i] = c0[i]; qa[8 + i] = c1[i]; }
  }

  float m[8], l[8];
  v8f acc[16];
  #pragma unroll
  for (int v = 0; v < 8; ++v) { m[v] = -__builtin_inff(); l[v] = 0.f; }
  #pragma unroll
  for (int ct = 0; ct < 16; ++ct)
    #pragma unroll
    for (int v = 0; v < 8; ++v) acc[ct][v] = 0.f;

  v8f zero;
  #pragma unroll
  for (int v = 0; v < 8; ++v) zero[v] = 0.f;

  issue_tile(0, 0);
  int cur = 0;
  constexpr int NIT = NPIX / 32;

  for (int it = 0; it < NIT; ++it) {
    __syncthreads();                         // buf cur^1 free to overwrite
    if (it + 1 < NIT) {
      issue_tile((it + 1) * 32, cur ^ 1);    // prefetch next tile (overlaps WMMAs)
      wait_async_le9();                      // previous 9 (this tile) have landed
    } else {
      wait_async_0();
    }
    __syncthreads();                         // all waves' chunks visible

    // key B-fragments from LDS
    v16h kf0, kf1;
    {
      const _Float16* r0 = &ckTile[cur][ln][hi * 16];
      const _Float16* r1 = &ckTile[cur][16 + ln][hi * 16];
      v8h a0 = *(const v8h*)r0;  v8h a1 = *(const v8h*)(r0 + 8);
      v8h b0 = *(const v8h*)r1;  v8h b1 = *(const v8h*)(r1 + 8);
      #pragma unroll
      for (int i = 0; i < 8; ++i) {
        kf0[i] = a0[i]; kf0[8 + i] = a1[i];
        kf1[i] = b0[i]; kf1[8 + i] = b1[i];
      }
    }
    v8f s0 = __builtin_amdgcn_wmma_f32_16x16x32_f16(false, qa, false, kf0, (short)0, zero, false, false);
    v8f s1 = __builtin_amdgcn_wmma_f32_16x16x32_f16(false, qa, false, kf1, (short)0, zero, false, false);

    // online softmax; row v+8*hi lives across the lane's own 16-lane group
    float p0[8], p1[8], scale[8];
    #pragma unroll
    for (int v = 0; v < 8; ++v) {
      float mv = fmaxf(s0[v], s1[v]);
      #pragma unroll
      for (int s = 1; s < 16; s <<= 1) mv = fmaxf(mv, __shfl_xor(mv, s, 16));
      float mn = fmaxf(m[v], mv);
      float sc = __expf(m[v] - mn);
      float e0 = __expf(s0[v] - mn);
      float e1 = __expf(s1[v] - mn);
      float rs = e0 + e1;
      #pragma unroll
      for (int s = 1; s < 16; s <<= 1) rs += __shfl_xor(rs, s, 16);
      l[v] = l[v] * sc + rs;
      m[v] = mn;
      p0[v] = e0; p1[v] = e1; scale[v] = sc;
    }
    #pragma unroll
    for (int ct = 0; ct < 16; ++ct)
      #pragma unroll
      for (int v = 0; v < 8; ++v) acc[ct][v] *= scale[v];

    // P (C/D layout) -> A layout through wave-local LDS
    #pragma unroll
    for (int v = 0; v < 8; ++v) {
      int row = v + hi * 8;
      ldsP[wid][row][ln]      = (_Float16)p0[v];
      ldsP[wid][row][16 + ln] = (_Float16)p1[v];
    }
    __builtin_amdgcn_wave_barrier();
    v16h pa;
    {
      int kb = hi * 16;
      const _Float16* r = &ldsP[wid][ln][0];
      v8h c0 = *(const v8h*)(r + kb);
      v8h c1 = *(const v8h*)(r + kb + 8);
      #pragma unroll
      for (int i = 0; i < 8; ++i) { pa[i] = c0[i]; pa[8 + i] = c1[i]; }
    }
    __builtin_amdgcn_wave_barrier();

    // acc += P(16x32) @ D(32x256), value fragments from LDS
    #pragma unroll 4
    for (int ct = 0; ct < 16; ++ct) {
      const _Float16* r = &dTile[cur][ct * 16 + ln][hi * 16];
      v8h d0 = *(const v8h*)r;  v8h d1 = *(const v8h*)(r + 8);
      v16h df;
      #pragma unroll
      for (int i = 0; i < 8; ++i) { df[i] = d0[i]; df[8 + i] = d1[i]; }
      acc[ct] = __builtin_amdgcn_wmma_f32_16x16x32_f16(false, pa, false, df,
                                                       (short)0, acc[ct], false, false);
    }
    cur ^= 1;
  }

  // epilogue: out = gamma * (acc / l) + x
  float g = gamma[0];
  #pragma unroll
  for (int ct = 0; ct < 16; ++ct) {
    #pragma unroll
    for (int v = 0; v < 8; ++v) {
      int row = q0 + v + hi * 8;
      size_t idx = ((size_t)(b * NPIX + row)) * CCH + ct * 16 + ln;
      out[idx] = g * (acc[ct][v] / l[v]) + x[idx];
    }
  }
}

// ---------------------------------------------------------------------------
extern "C" void kernel_launch(void* const* d_in, const int* in_sizes, int n_in,
                              void* d_out, int out_size, void* d_ws, size_t ws_size,
                              hipStream_t stream) {
  (void)in_sizes; (void)n_in; (void)out_size; (void)ws_size;
  const float* x  = (const float*)d_in[0];
  const float* Wb = (const float*)d_in[1];
  const float* bb = (const float*)d_in[2];
  const float* Wc = (const float*)d_in[3];
  const float* bc = (const float*)d_in[4];
  const float* Wd = (const float*)d_in[5];
  const float* bd = (const float*)d_in[6];
  const float* gm = (const float*)d_in[7];
  float* out = (float*)d_out;
  _Float16* ws = (_Float16*)d_ws;   // needs ~21.2 MB

  pam_pack_w<<<320, 256, 0, stream>>>(Wb, Wc, Wd, ws);
  pam_proj  <<<512, 128, 0, stream>>>(x, bb, bc, bd, ws);
  pam_attn  <<<512, 128, 0, stream>>>(x, gm, ws, out);
}
